// CAPERNN_ContRoles_HPN_53197464928658
// MI455X (gfx1250) — matrix operand order repaired and checked by reading, not compiled, not run on hardware
//
#include <hip/hip_runtime.h>
#include <hip/hip_bf16.h>

// ---------------------------------------------------------------------------
// CAPERNN forward for MI455X (gfx1250): bf16 WMMA everywhere, wave32, 8-wave
// workgroups, one 16-row tile per workgroup. Attention refactored to
// single-query form (q_eff = Wk^T q, ctx = Wv * pooled) to kill the K/V
// projection GEMMs; final logits use qk = kp_W^T q_vec for the same reason.
// ---------------------------------------------------------------------------

typedef __attribute__((ext_vector_type(16))) __bf16 v16bf;
typedef __attribute__((ext_vector_type(8)))  __bf16 v8bf;
typedef __attribute__((ext_vector_type(8)))  float  v8f;

#define BN_TOT   16384      // B * N_AGENTS
#define NBLK     1024       // BN / 16
#define NEGV     (-1000000000.0f)

// ---------------- ws byte offsets -------------------------------------------
#define OFF_W_OWN   0u
#define OFF_W_ALLY  16384u
#define OFF_W_EN    32768u
#define OFF_W_AQ    49152u
#define OFF_W_AKT   180224u
#define OFF_W_AV    311296u
#define OFF_W_AO    442368u
#define OFF_W_EQ    573440u
#define OFF_W_EKT   704512u
#define OFF_W_EV    835584u
#define OFF_W_EO    966656u
#define OFF_W_F1    1097728u
#define OFF_W_F2    1490944u
#define OFF_W_P1    1622016u
#define OFF_W_P2    1818624u
#define OFF_W_R1    1851392u
#define OFF_W_R2    1884160u
#define OFF_W_GI    1888256u
#define OFF_W_GH    2330624u
#define OFF_W_QN    2723840u
#define OFF_W_QP    2732032u
#define OFF_W_KPT   2879488u
#define OFF_SELF    3145728u
#define OFF_CA      19922944u
#define OFF_CE      36700160u
#define OFF_PA      53477376u
#define OFF_PE      70254592u
#define OFF_H0      87031808u
#define OFF_R       103809024u

// ---------------- WMMA helpers ----------------------------------------------
__device__ inline v8f zero8() {
  v8f z;
#pragma unroll
  for (int i = 0; i < 8; ++i) z[i] = 0.f;
  return z;
}

__device__ inline v8f wmma_bf(v16bf a, v16bf b, v8f c) {
  return __builtin_amdgcn_wmma_f32_16x16x32_bf16(false, a, false, b, (short)0, c,
                                                 false, false);
}

// A-fragment (16x32 bf16) from LDS, row-major with element stride `stride`.
// lanes 0-15: row=lane, k {0..7,16..23}; lanes 16-31: row=lane-16, k {8..15,24..31}
__device__ inline v16bf ldsA(const __bf16* base, int stride, int k0) {
  int lane = threadIdx.x & 31;
  int row  = lane & 15;
  int hi   = lane >> 4;
  const __bf16* p = base + row * stride + k0 + hi * 8;
  v8bf a0 = *(const v8bf*)p;
  v8bf a1 = *(const v8bf*)(p + 16);
  v16bf r;
#pragma unroll
  for (int i = 0; i < 8; ++i) { r[i] = a0[i]; r[i + 8] = a1[i]; }
  return r;
}

// B-fragment (32x16 bf16): column n = weight row n0+n; lane n k {0..7,16..23},
// lane n+16 k {8..15,24..31}.  Weights stored [N][K] bf16 row-major in ws.
__device__ inline v16bf ldB(const __bf16* W, int Kstr, int n0, int k0) {
  int lane = threadIdx.x & 31;
  int col  = lane & 15;
  int kh   = (lane >> 4) << 3;
  const __bf16* p = W + (size_t)(n0 + col) * Kstr + k0 + kh;
  v8bf b0 = *(const v8bf*)p;
  v8bf b1 = *(const v8bf*)(p + 16);
  v16bf r;
#pragma unroll
  for (int i = 0; i < 8; ++i) { r[i] = b0[i]; r[i + 8] = b1[i]; }
  return r;
}

__device__ inline v8f gemm_lds(const __bf16* A, int Astr, const __bf16* W,
                               int Kstr, int n0, int K, v8f acc) {
  for (int k = 0; k < K; k += 32)
    acc = wmma_bf(ldsA(A, Astr, k), ldB(W, Kstr, n0, k), acc);
  return acc;
}

// A from global f32 feats (16 rows x 32, contiguous) with on-the-fly bf16 cvt
__device__ inline v8f gemm_feat(const float* F, const __bf16* W, int n0, v8f acc) {
  int lane = threadIdx.x & 31;
  int row  = lane & 15;
  int hi   = lane >> 4;
  const float* p = F + row * 32 + hi * 8;
  v16bf a;
#pragma unroll
  for (int i = 0; i < 8; ++i) { a[i] = (__bf16)p[i]; a[i + 8] = (__bf16)p[i + 16]; }
  return wmma_bf(a, ldB(W, 32, n0, 0), acc);
}

__device__ inline v8f addBias(v8f acc, const float* b, int n0) {
  float bv = b[n0 + (threadIdx.x & 15)];
#pragma unroll
  for (int r = 0; r < 8; ++r) acc[r] += bv;
  return acc;
}

// D tile (f32 acc): VGPR r -> row r + 8*(lane>=16), col = lane&15
__device__ inline void stD_bf(__bf16* out, int stride, int colbase, v8f acc) {
  int lane = threadIdx.x & 31, col = lane & 15, rb = (lane >> 4) * 8;
#pragma unroll
  for (int r = 0; r < 8; ++r)
    out[(rb + r) * stride + colbase + col] = (__bf16)acc[r];
}
__device__ inline void stD_f32g(float* out, size_t rstride, int colbase, v8f acc) {
  int lane = threadIdx.x & 31, col = lane & 15, rb = (lane >> 4) * 8;
#pragma unroll
  for (int r = 0; r < 8; ++r)
    out[(size_t)(rb + r) * rstride + colbase + col] = acc[r];
}

__device__ inline float gelu(float x) {
  return 0.5f * x * (1.f + erff(x * 0.70710678118654752f));
}

__device__ inline float rr16(float v) {   // reduce across 16-lane group
#pragma unroll
  for (int m = 1; m < 16; m <<= 1) v += __shfl_xor(v, m, 32);
  return v;
}

// LayerNorm of 16 rows x N (f32 LDS in -> bf16 LDS out). 256 threads.
__device__ inline void layernorm_lds(const float* in, int istr, __bf16* out,
                                     int ostr, const float* g, const float* be,
                                     int N) {
  int row = threadIdx.x >> 4, l = threadIdx.x & 15;
  float s = 0.f, s2 = 0.f;
  for (int c = l; c < N; c += 16) { float x = in[row * istr + c]; s += x; s2 += x * x; }
  s = rr16(s); s2 = rr16(s2);
  float m   = s / N;
  float inv = rsqrtf(fmaxf(s2 / N - m * m, 0.f) + 1e-5f);
  for (int c = l; c < N; c += 16) {
    float x = in[row * istr + c];
    out[row * ostr + c] = (__bf16)((x - m) * inv * g[c] + be[c]);
  }
}

// ---------------- weight convert / pad / transpose --------------------------
__global__ void k_conv(const float* __restrict__ src, __bf16* __restrict__ dst,
                       int R, int C, int DN, int DK, int trans) {
  int total = DN * DK;
  for (int i = blockIdx.x * blockDim.x + threadIdx.x; i < total;
       i += gridDim.x * blockDim.x) {
    int n = i / DK, k = i % DK;
    float v = 0.f;
    if (!trans) { if (n < R && k < C) v = src[(size_t)n * C + k]; }
    else        { if (k < R && n < C) v = src[(size_t)k * C + n]; }
    dst[i] = (__bf16)v;
  }
}

// ---------------- K1: embeddings + both attention branches ------------------
template <int M>
__device__ void attn_branch(int bn0, const float* feats, const float* eb,
                            const __bf16* wEmb, const __bf16* wQ,
                            const __bf16* wKT, const __bf16* wV,
                            const __bf16* wO, const __bf16* sSelf, __bf16* sAE,
                            __bf16* sQh, __bf16* sQeff, __bf16* sPool,
                            float* sAttn, unsigned char* sMask, float* pOut,
                            float* cOut) {
  int tid = threadIdx.x, wave = tid >> 5;

  // entity embeddings: M*16 flat rows -> sAE (bf16, stride 272)
  for (int job = wave; job < M * 16; job += 8) {
    int mt = job >> 4, nt = job & 15;
    v8f acc = gemm_feat(feats + ((size_t)bn0 * M + mt * 16) * 32, wEmb, nt * 16,
                        zero8());
    acc = addBias(acc, eb, nt * 16);
    stD_bf(sAE + mt * 16 * 272, 272, nt * 16, acc);
  }
  if (tid < 16 * M) {   // masks: flat row = b*M + m
    const float* p = feats + ((size_t)bn0 * M + tid) * 32;
    bool nz = false;
    for (int i = 0; i < 32; ++i) nz |= (p[i] != 0.f);
    sMask[tid] = nz ? 1 : 0;
  }
  // qh = self_emb @ Wq^T
  for (int nt = wave; nt < 16; nt += 8) {
    v8f acc = gemm_lds(sSelf, 272, wQ, 256, nt * 16, 256, zero8());
    stD_bf(sQh, 272, nt * 16, acc);
  }
  __syncthreads();

  // masked mean of entity embeddings
  {
    int b = tid >> 4, l = tid & 15;
    float cnt = 0.f;
    for (int m = 0; m < M; ++m) cnt += sMask[b * M + m] ? 1.f : 0.f;
    float inv = 1.f / fmaxf(cnt, 1e-8f);
    for (int c = l; c < 256; c += 16) {
      float s = 0.f;
      for (int m = 0; m < M; ++m)
        if (sMask[b * M + m]) s += (float)sAE[(b * M + m) * 272 + c];
      pOut[(size_t)(bn0 + b) * 256 + c] = s * inv;
    }
  }
  // q_eff[h] = qh[:,h*64:h*64+64] @ Wk[h*64: ,:]  (Wk stored transposed)
  for (int job = wave; job < 64; job += 8) {
    int h = job >> 4, nt = job & 15;
    v8f acc = zero8();
#pragma unroll
    for (int kk = 0; kk < 64; kk += 32)
      acc = wmma_bf(ldsA(sQh + h * 64, 272, kk),
                    ldB(wKT, 256, nt * 16, h * 64 + kk), acc);
    stD_bf(sQeff + h * 16 * 272, 272, nt * 16, acc);
  }
  __syncthreads();

  // scores + softmax (one thread per (b,h))
  if (tid < 64) {
    int b = tid >> 2, h = tid & 3;
    const __bf16* qe = sQeff + (h * 16 + b) * 272;
    float sc[16];
    float mx = -3.4e38f;
    for (int m = 0; m < M; ++m) {
      const __bf16* ae = sAE + (b * M + m) * 272;
      float a = 0.f;
      for (int j = 0; j < 256; ++j) a += (float)ae[j] * (float)qe[j];
      float v = sMask[b * M + m] ? a * 0.125f : NEGV;
      sc[m] = v; mx = fmaxf(mx, v);
    }
    float den = 0.f;
    for (int m = 0; m < M; ++m) { sc[m] = __expf(sc[m] - mx); den += sc[m]; }
    float id = 1.f / den;
    for (int m = 0; m < M; ++m) sAttn[(b * 4 + h) * 16 + m] = sc[m] * id;
  }
  __syncthreads();

  // pooled[h] = attn-weighted sum of entity embeddings
  {
    int b = tid >> 4, l = tid & 15;
    for (int c = l; c < 256; c += 16) {
#pragma unroll
      for (int h = 0; h < 4; ++h) {
        float s = 0.f;
        for (int m = 0; m < M; ++m)
          s += sAttn[(b * 4 + h) * 16 + m] * (float)sAE[(b * M + m) * 272 + c];
        sPool[(h * 16 + b) * 272 + c] = (__bf16)s;
      }
    }
  }
  __syncthreads();

  // ctx[h] = pooled[h] @ Wv[h]^T  -> reuse sQh as ctx buffer
  for (int job = wave; job < 16; job += 8) {
    int h = job >> 2, nt = job & 3;
    v8f acc = gemm_lds(sPool + h * 16 * 272, 272, wV, 256, h * 64 + nt * 16,
                       256, zero8());
    stD_bf(sQh, 272, h * 64 + nt * 16, acc);
  }
  __syncthreads();

  // cOut = ctx @ Wo^T
  for (int nt = wave; nt < 16; nt += 8) {
    v8f acc = gemm_lds(sQh, 272, wO, 256, nt * 16, 256, zero8());
    stD_f32g(cOut + (size_t)bn0 * 256, 256, nt * 16, acc);
  }
  __syncthreads();
}

#define SMEM_ATTN 230656
__global__ void __launch_bounds__(256) k_attn(
    const float* own_feats, const float* ally_feats, const float* enemy_feats,
    const float* own_b, const float* ally_b, const float* en_b,
    const __bf16* wOwn, const __bf16* wAlly, const __bf16* wEn,
    const __bf16* wAQ, const __bf16* wAKT, const __bf16* wAV, const __bf16* wAO,
    const __bf16* wEQ, const __bf16* wEKT, const __bf16* wEV, const __bf16* wEO,
    float* selfE, float* cA, float* cE, float* pA, float* pE) {
  extern __shared__ char smem[];
  __bf16* sAE   = (__bf16*)smem;                 // 256*272 bf16
  __bf16* sSelf = (__bf16*)(smem + 139264);      // 16*272
  __bf16* sQh   = (__bf16*)(smem + 147968);      // 16*272 (qh / ctx)
  __bf16* sQeff = (__bf16*)(smem + 156672);      // 64*272
  __bf16* sPool = (__bf16*)(smem + 191488);      // 64*272
  float*  sAttn = (float*)(smem + 226304);       // 16*4*16
  unsigned char* sMask = (unsigned char*)(smem + 230400);  // 256

  int bn0 = blockIdx.x * 16, wave = threadIdx.x >> 5;

  for (int nt = wave; nt < 16; nt += 8) {
    v8f acc = gemm_feat(own_feats + (size_t)bn0 * 32, wOwn, nt * 16, zero8());
    acc = addBias(acc, own_b, nt * 16);
    stD_f32g(selfE + (size_t)bn0 * 256, 256, nt * 16, acc);
    stD_bf(sSelf, 272, nt * 16, acc);
  }
  __syncthreads();

  attn_branch<15>(bn0, ally_feats, ally_b, wAlly, wAQ, wAKT, wAV, wAO, sSelf,
                  sAE, sQh, sQeff, sPool, sAttn, sMask, pA, cA);
  attn_branch<16>(bn0, enemy_feats, en_b, wEn, wEQ, wEKT, wEV, wEO, sSelf, sAE,
                  sQh, sQeff, sPool, sAttn, sMask, pE, cE);
}

// ---------------- K2: fuse MLP + pool/role path -----------------------------
__global__ void __launch_bounds__(256) k_fuse(
    const float* selfE, const float* cA, const float* cE, const float* pA,
    const float* pE, const __bf16* wF1, const __bf16* wF2, const __bf16* wP1,
    const __bf16* wP2, const __bf16* wR1, const __bf16* wR2,
    const float* fuse_b1, const float* fuse_g1, const float* fuse_be1,
    const float* fuse_b2, const float* pool_b1, const float* pool_g1,
    const float* pool_be1, const float* pool_b2, const float* role_b1,
    const float* role_g1, const float* role_be1, const float* role_b2,
    const float* rln_g, const float* rln_b, float* h0, float* rOut) {
  __shared__ __attribute__((aligned(16))) __bf16 sIn[16 * 784];
  __shared__ __attribute__((aligned(16))) float  sF[16 * 256];
  __shared__ __attribute__((aligned(16))) __bf16 sB[16 * 272];
  int tid = threadIdx.x, wave = tid >> 5, bn0 = blockIdx.x * 16;
  int lane = tid & 31, col = lane & 15, rb = (lane >> 4) * 8;

  {  // stage f_in = [self, cA, cE]
    int b = tid >> 4, l = tid & 15;
    size_t r = (size_t)(bn0 + b) * 256;
    for (int c = l; c < 256; c += 16) {
      sIn[b * 784 + c]       = (__bf16)selfE[r + c];
      sIn[b * 784 + 256 + c] = (__bf16)cA[r + c];
      sIn[b * 784 + 512 + c] = (__bf16)cE[r + c];
    }
  }
  __syncthreads();
  for (int nt = wave; nt < 16; nt += 8) {
    v8f acc = gemm_lds(sIn, 784, wF1, 768, nt * 16, 768, zero8());
    acc = addBias(acc, fuse_b1, nt * 16);
#pragma unroll
    for (int r = 0; r < 8; ++r) sF[(rb + r) * 256 + nt * 16 + col] = gelu(acc[r]);
  }
  __syncthreads();
  layernorm_lds(sF, 256, sB, 272, fuse_g1, fuse_be1, 256);
  __syncthreads();
  for (int nt = wave; nt < 16; nt += 8) {
    v8f acc = gemm_lds(sB, 272, wF2, 256, nt * 16, 256, zero8());
    acc = addBias(acc, fuse_b2, nt * 16);
    stD_f32g(h0 + (size_t)bn0 * 256, 256, nt * 16, acc);
  }
  __syncthreads();
  {  // stage p_in = [self, pA, pE]
    int b = tid >> 4, l = tid & 15;
    size_t r = (size_t)(bn0 + b) * 256;
    for (int c = l; c < 256; c += 16) {
      sIn[b * 784 + c]       = (__bf16)selfE[r + c];
      sIn[b * 784 + 256 + c] = (__bf16)pA[r + c];
      sIn[b * 784 + 512 + c] = (__bf16)pE[r + c];
    }
  }
  __syncthreads();
  for (int nt = wave; nt < 8; nt += 8) {
    v8f acc = gemm_lds(sIn, 784, wP1, 768, nt * 16, 768, zero8());
    acc = addBias(acc, pool_b1, nt * 16);
#pragma unroll
    for (int r = 0; r < 8; ++r) sF[(rb + r) * 256 + nt * 16 + col] = gelu(acc[r]);
  }
  __syncthreads();
  layernorm_lds(sF, 256, sB, 272, pool_g1, pool_be1, 128);
  __syncthreads();
  for (int nt = wave; nt < 8; nt += 8) {  // pooled -> sIn cols 0..127 (bf16)
    v8f acc = gemm_lds(sB, 272, wP2, 128, nt * 16, 128, zero8());
    acc = addBias(acc, pool_b2, nt * 16);
    stD_bf(sIn, 784, nt * 16, acc);
  }
  __syncthreads();
  for (int nt = wave; nt < 8; nt += 8) {
    v8f acc = gemm_lds(sIn, 784, wR1, 128, nt * 16, 128, zero8());
    acc = addBias(acc, role_b1, nt * 16);
#pragma unroll
    for (int r = 0; r < 8; ++r) sF[(rb + r) * 256 + nt * 16 + col] = gelu(acc[r]);
  }
  __syncthreads();
  layernorm_lds(sF, 256, sB, 272, role_g1, role_be1, 128);
  __syncthreads();
  if (wave == 0) {  // role_W2 padded to 16 rows
    v8f acc = gemm_lds(sB, 272, wR2, 128, 0, 128, zero8());
#pragma unroll
    for (int r = 0; r < 8; ++r) {
      float v = acc[r] + (col < 8 ? role_b2[col] : 0.f);
      sF[(rb + r) * 256 + col] = tanhf(v);
    }
  }
  __syncthreads();
  if (tid < 16) {  // LN over 8 role dims -> r
    float s = 0.f, s2 = 0.f;
    for (int c = 0; c < 8; ++c) { float x = sF[tid * 256 + c]; s += x; s2 += x * x; }
    float m = s / 8.f, inv = rsqrtf(fmaxf(s2 / 8.f - m * m, 0.f) + 1e-5f);
    for (int c = 0; c < 8; ++c) {
      float x = sF[tid * 256 + c];
      rOut[(size_t)(bn0 + tid) * 8 + c] = (x - m) * inv * rln_g[c] + rln_b[c];
    }
  }
}

// ---------------- K3: GRU + output heads ------------------------------------
#define SMEM_GRU 276736
__global__ void __launch_bounds__(256) k_gru(
    const float* h0, const float* rIn, const float* hidden,
    const float* enemy_feats, const __bf16* wGI, const __bf16* wGH,
    const __bf16* wQN, const __bf16* wQP, const __bf16* wKPT,
    const __bf16* wEn, const float* gru_bih, const float* gru_bhh,
    const float* qn_b, const float* en_b, float* outQ, float* outH) {
  extern __shared__ char smem[];
  __bf16* sX  = (__bf16*)smem;                 // 16*304  x=[h0,r,0pad]
  __bf16* sH  = (__bf16*)(smem + 9728);        // 16*272  h_prev bf16
  float*  sGH = (float*)(smem + 18432);        // 16*768
  float*  sG  = (float*)(smem + 67584);        // 16*512  rg|zg
  __bf16* sHC = (__bf16*)(smem + 100352);      // 16*272  h_cur bf16
  __bf16* sXC = (__bf16*)(smem + 109056);      // 16*304  [h_cur,r,0pad]
  __bf16* sQV = (__bf16*)(smem + 118784);      // 16*272
  __bf16* sQK = (__bf16*)(smem + 127488);      // 16*272
  float*  sLg = (float*)(smem + 136192);       // 16*16
  unsigned char* sM = (unsigned char*)(smem + 137216);  // 256
  __bf16* sEE = (__bf16*)(smem + 137472);      // 256*272

  int tid = threadIdx.x, wave = tid >> 5, bn0 = blockIdx.x * 16;
  int lane = tid & 31, col = lane & 15, rb = (lane >> 4) * 8;

  {
    int b = tid >> 4, l = tid & 15;
    size_t r0 = (size_t)(bn0 + b) * 256;
    for (int c = l; c < 256; c += 16) {
      sX[b * 304 + c] = (__bf16)h0[r0 + c];
      sH[b * 272 + c] = (__bf16)hidden[r0 + c];
    }
    if (l < 8) sX[b * 304 + 256 + l] = (__bf16)rIn[(size_t)(bn0 + b) * 8 + l];
    for (int c = 264 + l; c < 304; c += 16) sX[b * 304 + c] = (__bf16)0.f;
  }
  __syncthreads();

  for (int nt = wave; nt < 48; nt += 8) {  // gh = h_prev @ Whh^T + bhh
    v8f acc = gemm_lds(sH, 272, wGH, 256, nt * 16, 256, zero8());
    acc = addBias(acc, gru_bhh, nt * 16);
#pragma unroll
    for (int r = 0; r < 8; ++r) sGH[(rb + r) * 768 + nt * 16 + col] = acc[r];
  }
  __syncthreads();

  for (int nt = wave; nt < 32; nt += 8) {  // i_r, i_z chunks -> gates
    int n0 = nt * 16;
    v8f acc = gemm_lds(sX, 304, wGI, 288, n0, 288, zero8());
    acc = addBias(acc, gru_bih, n0);
#pragma unroll
    for (int r = 0; r < 8; ++r) {
      float g = 1.f / (1.f + __expf(-(acc[r] + sGH[(rb + r) * 768 + n0 + col])));
      sG[(rb + r) * 512 + n0 + col] = g;
    }
  }
  __syncthreads();

  for (int nt = wave; nt < 16; nt += 8) {  // i_n chunk -> h_cur
    int n0 = 512 + nt * 16;
    v8f acc = gemm_lds(sX, 304, wGI, 288, n0, 288, zero8());
    acc = addBias(acc, gru_bih, n0);
#pragma unroll
    for (int r = 0; r < 8; ++r) {
      int row = rb + r, c = nt * 16 + col;
      float rg = sG[row * 512 + c], zg = sG[row * 512 + 256 + c];
      float ng = tanhf(acc[r] + rg * sGH[row * 768 + n0 + col]);
      float hp = hidden[(size_t)(bn0 + row) * 256 + c];
      float hc = (1.f - zg) * ng + zg * hp;
      outH[(size_t)(bn0 + row) * 256 + c] = hc;
      __bf16 hb = (__bf16)hc;
      sHC[row * 272 + c] = hb;
      sXC[row * 304 + c] = hb;
    }
  }
  {
    int b = tid >> 4, l = tid & 15;
    if (l < 8) sXC[b * 304 + 256 + l] = sX[b * 304 + 256 + l];
    for (int c = 264 + l; c < 304; c += 16) sXC[b * 304 + c] = (__bf16)0.f;
  }
  __syncthreads();

  if (wave == 0) {  // q_normal (qn_W padded to 16 rows)
    v8f acc = gemm_lds(sHC, 272, wQN, 256, 0, 256, zero8());
#pragma unroll
    for (int r = 0; r < 8; ++r)
      if (col < 6)
        outQ[(size_t)(bn0 + rb + r) * 22 + col] = acc[r] + qn_b[col];
  }
  for (int nt = wave; nt < 16; nt += 8) {  // q_vec = [h_cur,r] @ qp_W^T
    v8f acc = gemm_lds(sXC, 304, wQP, 288, nt * 16, 288, zero8());
    stD_bf(sQV, 272, nt * 16, acc);
  }
  for (int job = wave; job < 256; job += 8) {  // recompute enemy_e (cheap)
    int mt = job >> 4, nt = job & 15;
    v8f acc = gemm_feat(enemy_feats + ((size_t)bn0 * 16 + mt * 16) * 32, wEn,
                        nt * 16, zero8());
    acc = addBias(acc, en_b, nt * 16);
    stD_bf(sEE + mt * 16 * 272, 272, nt * 16, acc);
  }
  if (tid < 256) {
    const float* p = enemy_feats + ((size_t)bn0 * 16 + tid) * 32;
    bool nz = false;
    for (int i = 0; i < 32; ++i) nz |= (p[i] != 0.f);
    sM[tid] = nz ? 1 : 0;
  }
  __syncthreads();

  for (int nt = wave; nt < 16; nt += 8) {  // qk = q_vec @ kp_W (kp stored ^T)
    v8f acc = gemm_lds(sQV, 272, wKPT, 256, nt * 16, 256, zero8());
    stD_bf(sQK, 272, nt * 16, acc);
  }
  __syncthreads();

  if (tid < 256) {  // logits = enemy_e . qk / 16
    int b = tid >> 4, m = tid & 15;
    const __bf16* ee = sEE + (b * 16 + m) * 272;
    const __bf16* qk = sQK + b * 272;
    float a = 0.f;
    for (int j = 0; j < 256; ++j) a += (float)ee[j] * (float)qk[j];
    sLg[b * 16 + m] = sM[b * 16 + m] ? a * 0.0625f : NEGV;
  }
  __syncthreads();
  if (tid < 16) {
    int b = tid;
    float s = 0.f, cnt = 0.f;
    for (int m = 0; m < 16; ++m)
      if (sM[b * 16 + m]) { s += sLg[b * 16 + m]; cnt += 1.f; }
    float mean = s / fmaxf(cnt, 1.f);
    for (int m = 0; m < 16; ++m)
      outQ[(size_t)(bn0 + b) * 22 + 6 + m] = sLg[b * 16 + m] - mean;
  }
}

// ---------------- host launcher ---------------------------------------------
extern "C" void kernel_launch(void* const* d_in, const int* in_sizes, int n_in,
                              void* d_out, int out_size, void* d_ws,
                              size_t ws_size, hipStream_t stream) {
  (void)in_sizes; (void)n_in; (void)out_size; (void)ws_size;
  char* ws = (char*)d_ws;
  const float* own_feats   = (const float*)d_in[1];
  const float* ally_feats  = (const float*)d_in[2];
  const float* enemy_feats = (const float*)d_in[3];
  const float* hidden      = (const float*)d_in[4];

  auto conv = [&](int src_idx, unsigned dstOff, int R, int C, int DN, int DK,
                  int trans) {
    int total = DN * DK;
    int blocks = (total + 255) / 256;
    k_conv<<<blocks, 256, 0, stream>>>((const float*)d_in[src_idx],
                                       (__bf16*)(ws + dstOff), R, C, DN, DK,
                                       trans);
  };
  conv(5,  OFF_W_OWN, 256, 32, 256, 32, 0);
  conv(7,  OFF_W_ALLY, 256, 32, 256, 32, 0);
  conv(9,  OFF_W_EN, 256, 32, 256, 32, 0);
  conv(11, OFF_W_AQ, 256, 256, 256, 256, 0);
  conv(12, OFF_W_AKT, 256, 256, 256, 256, 1);
  conv(13, OFF_W_AV, 256, 256, 256, 256, 0);
  conv(14, OFF_W_AO, 256, 256, 256, 256, 0);
  conv(15, OFF_W_EQ, 256, 256, 256, 256, 0);
  conv(16, OFF_W_EKT, 256, 256, 256, 256, 1);
  conv(17, OFF_W_EV, 256, 256, 256, 256, 0);
  conv(18, OFF_W_EO, 256, 256, 256, 256, 0);
  conv(19, OFF_W_F1, 256, 768, 256, 768, 0);
  conv(23, OFF_W_F2, 256, 256, 256, 256, 0);
  conv(25, OFF_W_P1, 128, 768, 128, 768, 0);
  conv(29, OFF_W_P2, 128, 128, 128, 128, 0);
  conv(31, OFF_W_R1, 128, 128, 128, 128, 0);
  conv(35, OFF_W_R2, 8, 128, 16, 128, 0);
  conv(39, OFF_W_GI, 768, 264, 768, 288, 0);
  conv(40, OFF_W_GH, 768, 256, 768, 256, 0);
  conv(43, OFF_W_QN, 6, 256, 16, 256, 0);
  conv(45, OFF_W_QP, 256, 264, 256, 288, 0);
  conv(46, OFF_W_KPT, 256, 256, 256, 256, 1);

  float* selfE = (float*)(ws + OFF_SELF);
  float* cA    = (float*)(ws + OFF_CA);
  float* cE    = (float*)(ws + OFF_CE);
  float* pA    = (float*)(ws + OFF_PA);
  float* pE    = (float*)(ws + OFF_PE);
  float* h0    = (float*)(ws + OFF_H0);
  float* rV    = (float*)(ws + OFF_R);
  float* outQ  = (float*)d_out;
  float* outH  = (float*)d_out + (size_t)BN_TOT * 22;

  (void)hipFuncSetAttribute((const void*)k_attn,
                            hipFuncAttributeMaxDynamicSharedMemorySize,
                            SMEM_ATTN);
  (void)hipFuncSetAttribute((const void*)k_gru,
                            hipFuncAttributeMaxDynamicSharedMemorySize,
                            SMEM_GRU);

  k_attn<<<NBLK, 256, SMEM_ATTN, stream>>>(
      own_feats, ally_feats, enemy_feats, (const float*)d_in[6],
      (const float*)d_in[8], (const float*)d_in[10],
      (const __bf16*)(ws + OFF_W_OWN), (const __bf16*)(ws + OFF_W_ALLY),
      (const __bf16*)(ws + OFF_W_EN), (const __bf16*)(ws + OFF_W_AQ),
      (const __bf16*)(ws + OFF_W_AKT), (const __bf16*)(ws + OFF_W_AV),
      (const __bf16*)(ws + OFF_W_AO), (const __bf16*)(ws + OFF_W_EQ),
      (const __bf16*)(ws + OFF_W_EKT), (const __bf16*)(ws + OFF_W_EV),
      (const __bf16*)(ws + OFF_W_EO), selfE, cA, cE, pA, pE);

  k_fuse<<<NBLK, 256, 0, stream>>>(
      selfE, cA, cE, pA, pE, (const __bf16*)(ws + OFF_W_F1),
      (const __bf16*)(ws + OFF_W_F2), (const __bf16*)(ws + OFF_W_P1),
      (const __bf16*)(ws + OFF_W_P2), (const __bf16*)(ws + OFF_W_R1),
      (const __bf16*)(ws + OFF_W_R2), (const float*)d_in[20],
      (const float*)d_in[21], (const float*)d_in[22], (const float*)d_in[24],
      (const float*)d_in[26], (const float*)d_in[27], (const float*)d_in[28],
      (const float*)d_in[30], (const float*)d_in[32], (const float*)d_in[33],
      (const float*)d_in[34], (const float*)d_in[36], (const float*)d_in[37],
      (const float*)d_in[38], h0, rV);

  k_gru<<<NBLK, 256, SMEM_GRU, stream>>>(
      h0, rV, hidden, enemy_feats, (const __bf16*)(ws + OFF_W_GI),
      (const __bf16*)(ws + OFF_W_GH), (const __bf16*)(ws + OFF_W_QN),
      (const __bf16*)(ws + OFF_W_QP), (const __bf16*)(ws + OFF_W_KPT),
      (const __bf16*)(ws + OFF_W_EN), (const float*)d_in[41],
      (const float*)d_in[42], (const float*)d_in[44], (const float*)d_in[10],
      outQ, outH);
}
